// RelativeConfidenceCrossAttention_24902220382281
// MI455X (gfx1250) — compile-verified, matrix operand order checked by reading, not run on hardware
//
#include <hip/hip_runtime.h>
#include <hip/hip_bf16.h>

typedef __attribute__((ext_vector_type(8)))  __bf16 v8bf;
typedef __attribute__((ext_vector_type(16))) __bf16 v16bf;
typedef __attribute__((ext_vector_type(8)))  float  v8f;

#define WMMA_BF16(a, b, c) \
  __builtin_amdgcn_wmma_f32_16x16x32_bf16(false, (a), false, (b), (short)0, (c), false, false)

// ---------------------------------------------------------------------------
// A/B fragment loader for V_WMMA_F32_16X16X32_BF16.
// Per ISA layout: lane half h holds elements [8h..8h+8) and [16+8h..16+8h+8)
// of a 32-element row. Two 16-byte vector loads (global_load_b128/ds_load_b128).
// ---------------------------------------------------------------------------
__device__ __forceinline__ v16bf load_frag32(const __bf16* p, int h) {
  v8bf lo = *(const v8bf*)(p + 8 * h);
  v8bf hi = *(const v8bf*)(p + 16 + 8 * h);
  v16bf r;
#pragma unroll
  for (int i = 0; i < 8; ++i) { r[i] = lo[i]; r[i + 8] = hi[i]; }
  return r;
}

// Fire-and-forget 16B global -> LDS copy (ASYNCcnt-tracked). Inline asm keeps
// it portable across ROCm-7.2 / amdgpu-toolchain builtin arity differences.
__device__ __forceinline__ void async_g2l_b128(void* ldsPtr, const void* gPtr) {
  unsigned lds = (unsigned)(size_t)ldsPtr;               // low 32 bits = LDS offset
  unsigned long long ga = (unsigned long long)(size_t)gPtr;
  asm volatile("global_load_async_to_lds_b128 %0, %1, off"
               :: "v"(lds), "v"(ga) : "memory");
}

// ---------------------------------------------------------------------------
// fp32 -> bf16 cast (grid-stride)
// ---------------------------------------------------------------------------
__global__ __launch_bounds__(256) void cast_f32_bf16(const float* __restrict__ in,
                                                     __bf16* __restrict__ out, int n) {
  int i = blockIdx.x * blockDim.x + threadIdx.x;
  int stride = gridDim.x * blockDim.x;
  for (; i < n; i += stride) out[i] = (__bf16)in[i];
}

// ---------------------------------------------------------------------------
// Y = X @ W^T + bias with bf16 WMMA. One 16x16 output tile per wave.
//   mode 0: store bf16 at [B,H,N,D]      (Q, K)
//   mode 1: store bf16 at [B,H,D,N]      (V transposed for P*V B-fragments)
//   mode 2: store fp32 at [M, 768]       (output projection)
// ---------------------------------------------------------------------------
__global__ __launch_bounds__(128) void gemm_xwT(const __bf16* __restrict__ X,
                                                const __bf16* __restrict__ W,
                                                const float* __restrict__ bias,
                                                __bf16* __restrict__ outb,
                                                float* __restrict__ outf,
                                                int mode, int M) {
  const int K = 768;
  const int wave = threadIdx.x >> 5;
  const int lane = threadIdx.x & 31;
  const int h = lane >> 4, r = lane & 15;
  const int m0 = (blockIdx.x * 4 + wave) * 16;
  const int n0 = blockIdx.y * 16;
  if (m0 >= M) return;  // wave-uniform

  const __bf16* xrow = X + (size_t)(m0 + r) * K;
  const __bf16* wrow = W + (size_t)(n0 + r) * K;

  v8f acc = {};
#pragma unroll 4
  for (int kc = 0; kc < K; kc += 32) {
    v16bf a = load_frag32(xrow + kc, h);
    v16bf b = load_frag32(wrow + kc, h);
    acc = WMMA_BF16(a, b, acc);
  }

  const float bcol = bias[n0 + r];
#pragma unroll
  for (int j = 0; j < 8; ++j) {
    const int m = m0 + 8 * h + j;   // token row
    const int o = n0 + r;           // output channel
    const float val = acc[j] + bcol;
    if (mode == 2) {
      outf[(size_t)m * 768 + o] = val;
    } else {
      const int b_ = m >> 11, n_ = m & 2047;     // N = 2048
      const int head = o >> 6, d = o & 63;       // D = 64
      const size_t idx = (mode == 0)
          ? ((size_t)(b_ * 12 + head) * 2048 + n_) * 64 + d
          : ((size_t)(b_ * 12 + head) * 64 + d) * 2048 + n_;
      outb[idx] = (__bf16)val;
    }
  }
}

// ---------------------------------------------------------------------------
// Flash attention. One wave = 16 query rows x full head. 4 waves / block all
// share one head, so K/V tiles are staged once per block into LDS with
// async global->LDS copies (ASYNCcnt) and consumed by all 4 waves.
//  Q,K: [BH, N, D] bf16.  Vt: [BH, D, N] bf16 (transposed).
//  scores = QK^T/8 + (mask1[q]-mask2[k])/T ; online softmax; O = P V.
// ---------------------------------------------------------------------------
__global__ __launch_bounds__(128) void attn_kernel(const __bf16* __restrict__ Q,
                                                   const __bf16* __restrict__ Km,
                                                   const __bf16* __restrict__ Vt,
                                                   const float* __restrict__ mask1,
                                                   const float* __restrict__ mask2,
                                                   const float* __restrict__ temp,
                                                   __bf16* __restrict__ Ob) {
  __shared__ __bf16 kTile[32][64];    // 32 keys x 64 dims   (4 KB)
  __shared__ __bf16 vTile[64][32];    // 64 dims x 32 keys   (4 KB)
  __shared__ __bf16 ldsP[4][16][32];  // per-wave P staging  (4 KB)
  const int Nn = 2048, Dd = 64, Cc = 768, Hh = 12;

  const int tid  = threadIdx.x;
  const int wave = tid >> 5;
  const int lane = tid & 31;
  const int h = lane >> 4, r = lane & 15;
  const int bh = blockIdx.y;
  const int b_ = bh / Hh, head = bh % Hh;
  const int q0 = (blockIdx.x * 4 + wave) * 16;
  const float invT = 1.0f / temp[0];

  const __bf16* qp = Q + (size_t)bh * Nn * Dd;
  const __bf16* kp = Km + (size_t)bh * Nn * Dd;
  const __bf16* vp = Vt + (size_t)bh * Dd * Nn;

  // Q fragments for K-dim 0..31 and 32..63
  const v16bf qlo = load_frag32(qp + (size_t)(q0 + r) * Dd, h);
  const v16bf qhi = load_frag32(qp + (size_t)(q0 + r) * Dd + 32, h);

  float m1v[8];
#pragma unroll
  for (int j = 0; j < 8; ++j) m1v[j] = mask1[b_ * Nn + q0 + 8 * h + j];

  float mrun[8], lrun[8];
#pragma unroll
  for (int j = 0; j < 8; ++j) { mrun[j] = -3.0e38f; lrun[j] = 0.0f; }
  v8f acc[4];
#pragma unroll
  for (int t = 0; t < 4; ++t) acc[t] = (v8f){};

  for (int kb = 0; kb < Nn; kb += 32) {
    // ---- cooperative async staging of K/V tiles (once per block) ----
    // K tile: 32x64 bf16 = 256 16B chunks; 8 chunks per key row.
#pragma unroll
    for (int it = 0; it < 2; ++it) {
      const int c = tid + it * 128;
      const int row = c >> 3, col = (c & 7) * 8;
      async_g2l_b128(&kTile[row][col], kp + (size_t)(kb + row) * Dd + col);
    }
    // V tile: 64x32 bf16 = 256 16B chunks; 4 chunks per dim row.
#pragma unroll
    for (int it = 0; it < 2; ++it) {
      const int c = tid + it * 128;
      const int row = c >> 2, col = (c & 3) * 8;
      async_g2l_b128(&vTile[row][col], vp + (size_t)row * Nn + kb + col);
    }
    if (kb + 32 < Nn)  // speculative prefetch of next K tile into L2
      __builtin_prefetch(kp + (size_t)(kb + 32 + r) * Dd, 0, 1);
    asm volatile("s_wait_asynccnt 0x0" ::: "memory");
    __syncthreads();

    // ---- scores for keys [kb, kb+16) and [kb+16, kb+32) from LDS ----
    v8f s0 = {}, s1 = {};
    {
      v16bf k_lo = load_frag32(&kTile[r][0], h);
      v16bf k_hi = load_frag32(&kTile[r][32], h);
      s0 = WMMA_BF16(qlo, k_lo, s0);
      s0 = WMMA_BF16(qhi, k_hi, s0);
    }
    {
      v16bf k_lo = load_frag32(&kTile[16 + r][0], h);
      v16bf k_hi = load_frag32(&kTile[16 + r][32], h);
      s1 = WMMA_BF16(qlo, k_lo, s1);
      s1 = WMMA_BF16(qhi, k_hi, s1);
    }

    const float m2a = mask2[b_ * Nn + kb + r];
    const float m2b = mask2[b_ * Nn + kb + 16 + r];

    // online softmax; row j lives on 16 lanes of one half, so xor-reductions
    // with masks 1,2,4,8 stay within the half.
#pragma unroll
    for (int j = 0; j < 8; ++j) {
      float sa = s0[j] * 0.125f + (m1v[j] - m2a) * invT;
      float sb = s1[j] * 0.125f + (m1v[j] - m2b) * invT;
      float t = fmaxf(sa, sb);
      t = fmaxf(t, __shfl_xor(t, 1, 32));
      t = fmaxf(t, __shfl_xor(t, 2, 32));
      t = fmaxf(t, __shfl_xor(t, 4, 32));
      t = fmaxf(t, __shfl_xor(t, 8, 32));
      const float mnew = fmaxf(mrun[j], t);
      const float corr = __expf(mrun[j] - mnew);
      mrun[j] = mnew;
      const float p0 = __expf(sa - mnew);
      const float p1 = __expf(sb - mnew);
      float rs = p0 + p1;
      rs += __shfl_xor(rs, 1, 32);
      rs += __shfl_xor(rs, 2, 32);
      rs += __shfl_xor(rs, 4, 32);
      rs += __shfl_xor(rs, 8, 32);
      lrun[j] = lrun[j] * corr + rs;
#pragma unroll
      for (int t4 = 0; t4 < 4; ++t4) acc[t4][j] *= corr;
      // C-layout element (M=8h+j, N=r) -> LDS row-major 16x32 tile
      ldsP[wave][8 * h + j][r] = (__bf16)p0;
      ldsP[wave][8 * h + j][16 + r] = (__bf16)p1;
    }

    // per-wave LDS RAW: DS is in-order per wave; make it explicit.
    asm volatile("s_wait_dscnt 0x0" ::: "memory");

    // read P back in A-fragment layout
    const v16bf pf = load_frag32(&ldsP[wave][r][0], h);

    // O += P * V from LDS V tile
#pragma unroll
    for (int t4 = 0; t4 < 4; ++t4) {
      v16bf vf = load_frag32(&vTile[16 * t4 + r][0], h);
      acc[t4] = WMMA_BF16(pf, vf, acc[t4]);
    }

    __syncthreads();  // all waves done reading tiles before next staging
  }

  // normalize and store bf16 [B, N, C]
#pragma unroll
  for (int t4 = 0; t4 < 4; ++t4) {
#pragma unroll
    for (int j = 0; j < 8; ++j) {
      const int n_ = q0 + 8 * h + j;
      const int c_ = head * 64 + 16 * t4 + r;
      Ob[((size_t)(b_ * Nn + n_)) * Cc + c_] = (__bf16)(acc[t4][j] / lrun[j]);
    }
  }
}

// ---------------------------------------------------------------------------
// LayerNorm(0.5*x1 + y) * gamma + beta.  One block per row of 768.
// ---------------------------------------------------------------------------
__global__ __launch_bounds__(256) void ln_kernel(const float* __restrict__ x1,
                                                 const float* __restrict__ y,
                                                 const float* __restrict__ gamma,
                                                 const float* __restrict__ beta,
                                                 float* __restrict__ out) {
  const int Cc = 768;
  __shared__ float ssum[256], ssq[256];
  const int row = blockIdx.x;
  const float* xr = x1 + (size_t)row * Cc;
  const float* yr = y + (size_t)row * Cc;

  float v[3], s = 0.0f, sq = 0.0f;
#pragma unroll
  for (int i = 0; i < 3; ++i) {
    const int c = threadIdx.x + 256 * i;
    const float z = 0.5f * xr[c] + yr[c];
    v[i] = z; s += z; sq += z * z;
  }
  ssum[threadIdx.x] = s; ssq[threadIdx.x] = sq;
  __syncthreads();
  for (int off = 128; off > 0; off >>= 1) {
    if (threadIdx.x < off) {
      ssum[threadIdx.x] += ssum[threadIdx.x + off];
      ssq[threadIdx.x] += ssq[threadIdx.x + off];
    }
    __syncthreads();
  }
  const float mu = ssum[0] * (1.0f / Cc);
  const float var = ssq[0] * (1.0f / Cc) - mu * mu;
  const float rstd = rsqrtf(var + 1e-5f);
#pragma unroll
  for (int i = 0; i < 3; ++i) {
    const int c = threadIdx.x + 256 * i;
    out[(size_t)row * Cc + c] = (v[i] - mu) * rstd * gamma[c] + beta[c];
  }
}

// ---------------------------------------------------------------------------
extern "C" void kernel_launch(void* const* d_in, const int* in_sizes, int n_in,
                              void* d_out, int out_size, void* d_ws, size_t ws_size,
                              hipStream_t stream) {
  (void)in_sizes; (void)n_in; (void)out_size; (void)ws_size;

  const float* x1    = (const float*)d_in[0];
  const float* x2    = (const float*)d_in[1];
  const float* mask1 = (const float*)d_in[2];
  const float* mask2 = (const float*)d_in[3];
  const float* Wq    = (const float*)d_in[4];
  const float* bq    = (const float*)d_in[5];
  const float* Wk    = (const float*)d_in[6];
  const float* bk    = (const float*)d_in[7];
  const float* Wv    = (const float*)d_in[8];
  const float* bv    = (const float*)d_in[9];
  const float* Wo    = (const float*)d_in[10];
  const float* bo    = (const float*)d_in[11];
  const float* temp  = (const float*)d_in[12];
  const float* gamma = (const float*)d_in[13];
  const float* beta  = (const float*)d_in[14];
  float* out = (float*)d_out;

  const int Bb = 4, Nn = 2048, Cc = 768;
  const int M = Bb * Nn;  // 8192
  const int BH = 48;

  // workspace carve-up (~105 MB total)
  char* ws = (char*)d_ws;
  size_t off = 0;
  auto alloc = [&](size_t bytes) -> void* {
    void* p = ws + off;
    off = (off + bytes + 255) & ~(size_t)255;
    return p;
  };
  __bf16* x1b  = (__bf16*)alloc((size_t)M * Cc * 2);
  __bf16* x2b  = (__bf16*)alloc((size_t)M * Cc * 2);
  __bf16* wqb  = (__bf16*)alloc((size_t)Cc * Cc * 2);
  __bf16* wkb  = (__bf16*)alloc((size_t)Cc * Cc * 2);
  __bf16* wvb  = (__bf16*)alloc((size_t)Cc * Cc * 2);
  __bf16* wob  = (__bf16*)alloc((size_t)Cc * Cc * 2);
  __bf16* qb   = (__bf16*)alloc((size_t)BH * Nn * 64 * 2);
  __bf16* kbuf = (__bf16*)alloc((size_t)BH * Nn * 64 * 2);
  __bf16* vtb  = (__bf16*)alloc((size_t)BH * 64 * Nn * 2);
  __bf16* attb = (__bf16*)alloc((size_t)M * Cc * 2);
  float*  projf = (float*)alloc((size_t)M * Cc * 4);

  // 1) casts to bf16
  cast_f32_bf16<<<2048, 256, 0, stream>>>(x1, x1b, M * Cc);
  cast_f32_bf16<<<2048, 256, 0, stream>>>(x2, x2b, M * Cc);
  cast_f32_bf16<<<512, 256, 0, stream>>>(Wq, wqb, Cc * Cc);
  cast_f32_bf16<<<512, 256, 0, stream>>>(Wk, wkb, Cc * Cc);
  cast_f32_bf16<<<512, 256, 0, stream>>>(Wv, wvb, Cc * Cc);
  cast_f32_bf16<<<512, 256, 0, stream>>>(Wo, wob, Cc * Cc);

  // 2) QKV projections (WMMA). grid = (M/64, 768/16), 4 waves/block.
  dim3 ggrid(M / 64, Cc / 16);
  gemm_xwT<<<ggrid, 128, 0, stream>>>(x1b, wqb, bq, qb,   nullptr, 0, M);
  gemm_xwT<<<ggrid, 128, 0, stream>>>(x2b, wkb, bk, kbuf, nullptr, 0, M);
  gemm_xwT<<<ggrid, 128, 0, stream>>>(x2b, wvb, bv, vtb,  nullptr, 1, M);  // transposed

  // 3) flash attention: 128 query tiles x 48 heads, 4 waves/block
  dim3 agrid(Nn / 64, BH);
  attn_kernel<<<agrid, 128, 0, stream>>>(qb, kbuf, vtb, mask1, mask2, temp, attb);

  // 4) output projection (fp32 result)
  gemm_xwT<<<ggrid, 128, 0, stream>>>(attb, wob, bo, nullptr, projf, 2, M);

  // 5) residual + LayerNorm -> d_out
  ln_kernel<<<M, 256, 0, stream>>>(x1, projf, gamma, beta, out);
}
